// PointnetSAModuleVotes_5669356832929
// MI455X (gfx1250) — matrix operand order repaired and checked by reading, not compiled, not run on hardware
//
#include <hip/hip_runtime.h>
#include <hip/hip_bf16.h>

typedef __attribute__((ext_vector_type(16))) _Float16 v16h;
typedef __attribute__((ext_vector_type(8)))  _Float16 v8h;
typedef __attribute__((ext_vector_type(8)))  float    v8f;

#define NPTS    16384
#define NPOINT  1024
#define NSAMPLE 64
#define BATCH   8
#define RADIUS  0.2f

#define NFRAG   28            // 4 (L1) + 8 (L2) + 16 (L3) B-matrix fragments
#define STRA    72            // activation row stride in halves (padded)

struct MlpArgs {
  const float* W[3];
  const float* bb[3];
  const float* ga[3];
  const float* be[3];
  const float* mu[3];
  const float* va[3];
};

// ---------------------------------------------------------------------------
// Kernel 0: one-time prep. Repack weights into fragment-native f16 layout
// (each lane's 16 K-halves of one output channel, two 8-runs at 8h and 16+8h), and fold BN
// (scale = gamma*rsqrt(var+eps), bias = (b-mean)*scale + beta).
// Fragment order: L1 frags 0..3 (nt, ks=0, cin=4 zero-padded to 32),
//                 L2 frags 4..11 (nt*2+ks), L3 frags 12..27 (nt*2+ks).
// ---------------------------------------------------------------------------
__global__ __launch_bounds__(256) void prep_kernel(MlpArgs p,
    _Float16* __restrict__ wsW, float* __restrict__ wsSB)
{
  const int tid = threadIdx.x;
  for (int e = tid; e < NFRAG * 32; e += 256) {
    const int f = e >> 5, lane = e & 31;
    int nt, ks, cin; const float* W;
    if (f < 4)       { nt = f;            ks = 0;              cin = 4;  W = p.W[0]; }
    else if (f < 12) { int q = f - 4;  nt = q >> 1; ks = q & 1; cin = 64; W = p.W[1]; }
    else             { int q = f - 12; nt = q >> 1; ks = q & 1; cin = 64; W = p.W[2]; }
    const int col = nt * 16 + (lane & 15);
    const int k0  = ks * 32 + ((lane >> 4) << 3);
    _Float16* dst = wsW + (size_t)e * 16;
#pragma unroll
    for (int h = 0; h < 16; ++h) {
      int K = k0 + (h < 8 ? h : h + 8);
      dst[h] = (_Float16)((K < cin) ? W[col * cin + K] : 0.0f);
    }
  }
  if (tid < 64) {
    float sc = p.ga[0][tid] * rsqrtf(p.va[0][tid] + 1e-5f);
    wsSB[tid]       = sc;
    wsSB[64 + tid]  = (p.bb[0][tid] - p.mu[0][tid]) * sc + p.be[0][tid];
    sc = p.ga[1][tid] * rsqrtf(p.va[1][tid] + 1e-5f);
    wsSB[128 + tid] = sc;
    wsSB[192 + tid] = (p.bb[1][tid] - p.mu[1][tid]) * sc + p.be[1][tid];
  }
  if (tid < 128) {
    float sc = p.ga[2][tid] * rsqrtf(p.va[2][tid] + 1e-5f);
    wsSB[256 + tid] = sc;
    wsSB[384 + tid] = (p.bb[2][tid] - p.mu[2][tid]) * sc + p.be[2][tid];
  }
}

// ---------------------------------------------------------------------------
// Kernel 1: farthest point sampling. One 1024-thread block per batch; each
// thread owns 16 points in registers; argmax via wave shuffles + LDS tree.
// ---------------------------------------------------------------------------
__global__ __launch_bounds__(1024) void fps_kernel(
    const float* __restrict__ xyz, float* __restrict__ out_xyz,
    float* __restrict__ out_inds_f)
{
  const int b    = blockIdx.x;
  const int tid  = threadIdx.x;
  const int lane = tid & 31;
  const int wv   = tid >> 5;
  const float* base = xyz + (size_t)b * NPTS * 3;

  constexpr int PPT = NPTS / 1024;
  float px[PPT], py[PPT], pz[PPT], dist[PPT];
#pragma unroll
  for (int i = 0; i < PPT; ++i) {
    int p = tid + i * 1024;
    px[i] = base[p * 3 + 0];
    py[i] = base[p * 3 + 1];
    pz[i] = base[p * 3 + 2];
    dist[i] = 1e10f;
  }

  __shared__ float sx, sy, sz;
  __shared__ float swD[32];
  __shared__ int   swI[32];

  if (tid == 0) {
    sx = base[0]; sy = base[1]; sz = base[2];
    out_inds_f[b * NPOINT] = 0.0f;
    out_xyz[(size_t)(b * NPOINT) * 3 + 0] = sx;
    out_xyz[(size_t)(b * NPOINT) * 3 + 1] = sy;
    out_xyz[(size_t)(b * NPOINT) * 3 + 2] = sz;
  }
  __syncthreads();

  for (int s = 1; s < NPOINT; ++s) {
    const float cx = sx, cy = sy, cz = sz;
    float best = -1.0f; int bi = 0;
#pragma unroll
    for (int i = 0; i < PPT; ++i) {
      float dx = px[i] - cx, dy = py[i] - cy, dz = pz[i] - cz;
      float nd = dx * dx + dy * dy + dz * dz;
      float d  = dist[i];
      d = nd < d ? nd : d;
      dist[i] = d;
      if (d > best) { best = d; bi = tid + i * 1024; }
    }
    // wave32 argmax (lowest index on ties, matching jnp.argmax)
#pragma unroll
    for (int off = 16; off > 0; off >>= 1) {
      float ob = __shfl_down(best, off, 32);
      int   oi = __shfl_down(bi,   off, 32);
      if (ob > best || (ob == best && oi < bi)) { best = ob; bi = oi; }
    }
    if (lane == 0) { swD[wv] = best; swI[wv] = bi; }
    __syncthreads();
    if (wv == 0) {
      best = swD[lane]; bi = swI[lane];
#pragma unroll
      for (int off = 16; off > 0; off >>= 1) {
        float ob = __shfl_down(best, off, 32);
        int   oi = __shfl_down(bi,   off, 32);
        if (ob > best || (ob == best && oi < bi)) { best = ob; bi = oi; }
      }
      if (lane == 0) {
        sx = base[bi * 3 + 0]; sy = base[bi * 3 + 1]; sz = base[bi * 3 + 2];
        int g = b * NPOINT + s;
        out_inds_f[g] = (float)bi;
        out_xyz[(size_t)g * 3 + 0] = sx;
        out_xyz[(size_t)g * 3 + 1] = sy;
        out_xyz[(size_t)g * 3 + 2] = sz;
      }
    }
    __syncthreads();
  }
}

// ---------------------------------------------------------------------------
// Kernel 2: ball query. One wave per query point; ordered index compaction
// via ballot + prefix popcount (keeps the "64 smallest indices" semantics),
// early exit once 64 found; pad with first index.
// ---------------------------------------------------------------------------
__global__ __launch_bounds__(256) void ball_query_kernel(
    const float* __restrict__ xyz, const float* __restrict__ new_xyz,
    int* __restrict__ ws_ball)
{
  const int wave = (int)((blockIdx.x * blockDim.x + threadIdx.x) >> 5);
  const int lane = threadIdx.x & 31;
  const int b = wave >> 10;
  const float r2 = RADIUS * RADIUS;

  const float cx = new_xyz[(size_t)wave * 3 + 0];
  const float cy = new_xyz[(size_t)wave * 3 + 1];
  const float cz = new_xyz[(size_t)wave * 3 + 2];
  const float* base = xyz + (size_t)b * NPTS * 3;
  int* dst = ws_ball + (size_t)wave * NSAMPLE;

  int cnt = 0;
  for (int start = 0; start < NPTS && cnt < NSAMPLE; start += 32) {
    int i = start + lane;
    float dx = base[i * 3 + 0] - cx;
    float dy = base[i * 3 + 1] - cy;
    float dz = base[i * 3 + 2] - cz;
    bool in = (dx * dx + dy * dy + dz * dz) < r2;
    unsigned mask = (unsigned)__ballot(in);
    if (in) {
      int pos = cnt + __popc(mask & ((1u << lane) - 1u));
      if (pos < NSAMPLE) dst[pos] = i;
    }
    cnt += __popc(mask);
  }
  if (cnt > NSAMPLE) cnt = NSAMPLE;
  int first = dst[0];  // query point itself is always in-radius -> cnt >= 1
  for (int j = cnt + lane; j < NSAMPLE; j += 32) dst[j] = first;
}

// ---------------------------------------------------------------------------
// Kernel 3: gather + 3-layer MLP (f16 WMMA 16x16x32) + fused max-pool.
// One workgroup (4 waves) per query point. Only activations live in LDS;
// B fragments stream from the prepacked global table straight to registers.
// ---------------------------------------------------------------------------

// A fragment (16-bit A 16x32, ISA 7.12.2): per lane two contiguous 8-half
// runs at K = kbase + (lane>=16 ? 8 : 0) and that +16.
__device__ __forceinline__ v16h load_frag_A(const _Float16* base, int lane,
                                            int kbase) {
  const _Float16* ptr = base + (lane & 15) * STRA + kbase + ((lane >> 4) << 3);
  union { v16h v; v8h h[2]; } u;
  u.h[0] = *(const v8h*)(ptr);
  u.h[1] = *(const v8h*)(ptr + 16);
  return u.v;
}

// B fragment from the fragment-native global table: lane's 16 halves are
// contiguous -> two coalesced global b128 loads.
__device__ __forceinline__ v16h load_frag_W(const _Float16* __restrict__ wsW,
                                            int frag, int lane) {
  const _Float16* ptr = wsW + ((size_t)(frag * 32 + lane)) * 16;
  union { v16h v; v8h h[2]; } u;
  u.h[0] = *(const v8h*)(ptr);
  u.h[1] = *(const v8h*)(ptr + 8);
  return u.v;
}

__device__ __forceinline__ void mlp_layer(const _Float16* in_act, _Float16* out_act,
    const _Float16* __restrict__ wsW, int fragBase, int ksteps, int ntiles,
    const float* __restrict__ sc, const float* __restrict__ bi,
    int mt, int lane)
{
  // Hoist A fragments: reused by every output tile of this layer.
  v16h a[2];
  const _Float16* abase = in_act + mt * 16 * STRA;
  for (int ks = 0; ks < ksteps; ++ks) a[ks] = load_frag_A(abase, lane, ks * 32);

  for (int nt = 0; nt < ntiles; ++nt) {
    v8f acc = {};
    for (int ks = 0; ks < ksteps; ++ks) {
      v16h bm = load_frag_W(wsW, fragBase + nt * ksteps + ks, lane);
      acc = __builtin_amdgcn_wmma_f32_16x16x32_f16(false, a[ks], false, bm,
                                                   (short)0, acc, false, false);
    }
    const int col = nt * 16 + (lane & 15);
    const float scale = sc[col], bias = bi[col];
    const int rbase = mt * 16 + ((lane >> 4) << 3);
#pragma unroll
    for (int v = 0; v < 8; ++v) {
      float y = acc[v] * scale + bias;             // folded linear-bias + BN
      out_act[(rbase + v) * STRA + col] = (_Float16)(y > 0.0f ? y : 0.0f);
    }
  }
}

__global__ __launch_bounds__(128) void group_mlp_kernel(
    const float* __restrict__ xyz, const float* __restrict__ features,
    const float* __restrict__ new_xyz, const int* __restrict__ ws_ball,
    const _Float16* __restrict__ wsW, const float* __restrict__ wsSB,
    float* __restrict__ out_feat)
{
  __shared__ __align__(16) _Float16 actA[64 * STRA];
  __shared__ __align__(16) _Float16 actB[64 * STRA];
  __shared__ float sPart[4 * 128];

  const int tid  = threadIdx.x;
  const int lane = tid & 31;
  const int wv   = tid >> 5;
  const int g = blockIdx.x;
  const int b = g >> 10;
  const int s = g & (NPOINT - 1);

  // --- gather X0 = [ (xyz - center)/R , feature ] padded to K=32 ---
  const float cx = new_xyz[(size_t)g * 3 + 0];
  const float cy = new_xyz[(size_t)g * 3 + 1];
  const float cz = new_xyz[(size_t)g * 3 + 2];
  if (tid < NSAMPLE) {
    int idx = ws_ball[(size_t)g * NSAMPLE + tid];
    const float* pp = xyz + ((size_t)b * NPTS + idx) * 3;
    float f = features[(size_t)b * NPTS + idx];
    _Float16* row = actA + tid * STRA;
    row[0] = (_Float16)((pp[0] - cx) * (1.0f / RADIUS));
    row[1] = (_Float16)((pp[1] - cy) * (1.0f / RADIUS));
    row[2] = (_Float16)((pp[2] - cz) * (1.0f / RADIUS));
    row[3] = (_Float16)f;
#pragma unroll
    for (int c = 4; c < 32; ++c) row[c] = (_Float16)0.0f;
  }
  __syncthreads();

  // layer 1: 64x32(pad) x 32x64 -> actB
  mlp_layer(actA, actB, wsW, 0, 1, 4, wsSB + 0, wsSB + 64, wv, lane);
  __syncthreads();
  // layer 2: 64x64 x 64x64 -> actA
  mlp_layer(actB, actA, wsW, 4, 2, 4, wsSB + 128, wsSB + 192, wv, lane);
  __syncthreads();

  // layer 3: 64x64 x 64x128, fused max over the 64 samples
  {
    const _Float16* abase = actA + wv * 16 * STRA;
    v16h a0 = load_frag_A(abase, lane, 0);
    v16h a1 = load_frag_A(abase, lane, 32);
    for (int nt = 0; nt < 8; ++nt) {
      v8f acc = {};
      acc = __builtin_amdgcn_wmma_f32_16x16x32_f16(false, a0, false,
              load_frag_W(wsW, 12 + nt * 2 + 0, lane), (short)0, acc, false, false);
      acc = __builtin_amdgcn_wmma_f32_16x16x32_f16(false, a1, false,
              load_frag_W(wsW, 12 + nt * 2 + 1, lane), (short)0, acc, false, false);
      const int col = nt * 16 + (lane & 15);
      const float sc = wsSB[256 + col], bi = wsSB[384 + col];
      float mx = 0.0f;  // ReLU outputs are >= 0
#pragma unroll
      for (int v = 0; v < 8; ++v) {
        float y = acc[v] * sc + bi;
        y = y > 0.0f ? y : 0.0f;
        mx = y > mx ? y : mx;
      }
      float o = __shfl_xor(mx, 16, 32);     // lanes n and n+16 share a column
      mx = o > mx ? o : mx;
      if (lane < 16) sPart[wv * 128 + col] = mx;
    }
  }
  __syncthreads();
  if (tid < 128) {
    float m0 = sPart[tid],       m1 = sPart[128 + tid];
    float m2 = sPart[256 + tid], m3 = sPart[384 + tid];
    float m = fmaxf(fmaxf(m0, m1), fmaxf(m2, m3));
    out_feat[((size_t)b * 128 + tid) * NPOINT + s] = m;
  }
}

// ---------------------------------------------------------------------------
extern "C" void kernel_launch(void* const* d_in, const int* in_sizes, int n_in,
                              void* d_out, int out_size, void* d_ws, size_t ws_size,
                              hipStream_t stream) {
  const float* xyz      = (const float*)d_in[0];
  const float* features = (const float*)d_in[1];
  MlpArgs p;
  for (int l = 0; l < 3; ++l) {
    p.W[l]  = (const float*)d_in[2 + 6 * l + 0];
    p.bb[l] = (const float*)d_in[2 + 6 * l + 1];
    p.ga[l] = (const float*)d_in[2 + 6 * l + 2];
    p.be[l] = (const float*)d_in[2 + 6 * l + 3];
    p.mu[l] = (const float*)d_in[2 + 6 * l + 4];
    p.va[l] = (const float*)d_in[2 + 6 * l + 5];
  }

  float* out      = (float*)d_out;
  float* out_xyz  = out;                                              // 8*1024*3
  float* out_feat = out + BATCH * NPOINT * 3;                         // 8*128*1024
  float* out_inds = out + BATCH * NPOINT * 3 + BATCH * 128 * NPOINT;  // 8*1024

  // workspace: [ ball idx : 8*1024*64 int ][ wsW : 28*32*16 f16 ][ wsSB : 512 f32 ]
  char* ws = (char*)d_ws;
  int*       ws_ball = (int*)ws;
  _Float16*  wsW     = (_Float16*)(ws + (size_t)BATCH * NPOINT * NSAMPLE * 4);
  float*     wsSB    = (float*)(ws + (size_t)BATCH * NPOINT * NSAMPLE * 4
                                   + (size_t)NFRAG * 32 * 16 * 2);

  prep_kernel<<<1, 256, 0, stream>>>(p, wsW, wsSB);
  fps_kernel<<<BATCH, 1024, 0, stream>>>(xyz, out_xyz, out_inds);
  ball_query_kernel<<<(BATCH * NPOINT * 32) / 256, 256, 0, stream>>>(xyz, out_xyz, ws_ball);
  group_mlp_kernel<<<BATCH * NPOINT, 128, 0, stream>>>(xyz, features, out_xyz,
                                                       ws_ball, wsW, wsSB, out_feat);
}
